// SeqNet_45930380263771
// MI455X (gfx1250) — compile-verified, matrix operand order checked by reading, not run on hardware
//
#include <hip/hip_runtime.h>

// ---------------------------------------------------------------------------
// SNN forward for MI455X (gfx1250, wave32, WMMA).
// Prep   : W0 fp32 -> bf16, zero-padded to [208][800] in scratch.
// Phase 1: Z[T*B, H] = X @ W0^T + b0 via v_wmma_f32_16x16x32_bf16,
//          B staged to LDS with async-to-LDS copies (double-buffered).
// Phase 2: per-batch LIF/LI scan over T=100 steps (VALU + LDS reduce).
// ---------------------------------------------------------------------------

typedef __bf16 bf16_t;
typedef __attribute__((ext_vector_type(16))) bf16_t v16bf;
typedef __attribute__((ext_vector_type(8)))  float  v8f;
typedef __attribute__((ext_vector_type(4)))  float  v4f;
typedef __attribute__((ext_vector_type(8)))  unsigned short ushort8;
typedef __attribute__((vector_size(16)))     int    v4i_t;
typedef __attribute__((address_space(1)))    v4i_t  gas1_v4i;   // global b128
typedef __attribute__((address_space(3)))    v4i_t  las3_v4i;   // LDS b128

#define SNN_T 100
#define SNN_B 1024
#define SNN_F 784
#define SNN_H 200
#define SNN_O 10
#define MROWS   (SNN_T * SNN_B)   // 102400 GEMM rows
#define MTILES  (MROWS / 16)      // 6400 M-tiles of 16
#define NTILES  13                // ceil(200/16)
#define KCHUNKS 25                // ceil(784/32)
#define LDSB_STRIDE 40            // halves per LDS B row: 80B (16B aligned, bank-rotating)
#define NROWS_PAD 208             // H padded to 13*16
#define KPAD 800                  // F padded to 25*32
#define NSEG (NROWS_PAD * 4)      // 832 x 16B segments per K-chunk

#if defined(__has_builtin)
#if __has_builtin(__builtin_amdgcn_global_load_async_to_lds_b128)
#define HAVE_ASYNC_LDS 1
#endif
#endif
#ifndef HAVE_ASYNC_LDS
#define HAVE_ASYNC_LDS 0
#endif

__device__ __forceinline__ unsigned short f2bfbits(float f) {
  // round-to-nearest-even fp32 -> bf16 bit pattern
  unsigned u = __float_as_uint(f);
  u += 0x7FFFu + ((u >> 16) & 1u);
  return (unsigned short)(u >> 16);
}

// ---------------------------------------------------------------------------
// Prep: W0[200,784] fp32 -> W0bf[208,800] bf16, zero padded.
// ---------------------------------------------------------------------------
__global__ __launch_bounds__(256) void snn_w0_prep(
    const float* __restrict__ W0, unsigned short* __restrict__ W0bf) {
  const int idx = blockIdx.x * 256 + threadIdx.x;
  if (idx >= NROWS_PAD * KPAD) return;
  const int n = idx / KPAD, k = idx - n * KPAD;
  unsigned short h = 0;
  if (n < SNN_H && k < SNN_F) h = f2bfbits(W0[n * SNN_F + k]);
  W0bf[idx] = h;
}

// ---------------------------------------------------------------------------
// Stage one 32-K-wide B chunk (208 rows x 64B) into an LDS buffer.
// Async path: global_load_async_to_lds_b128 (ASYNCcnt). Fallback: b128
// load + ds_store. Source is pre-padded bf16, so no guards/conversion.
// ---------------------------------------------------------------------------
__device__ __forceinline__ void stage_b_chunk(
    const unsigned short* __restrict__ W0bf, unsigned short* lds_buf,
    int kbase, int tid) {
  for (int id = tid; id < NSEG; id += 256) {
    const int n = id >> 2;
    const int seg = (id & 3) * 8;                   // half-element offset
    const unsigned short* src = W0bf + n * KPAD + kbase + seg;
    unsigned short* dst = lds_buf + n * LDSB_STRIDE + seg;
#if HAVE_ASYNC_LDS
    __builtin_amdgcn_global_load_async_to_lds_b128(
        (gas1_v4i*)const_cast<unsigned short*>(src),
        (las3_v4i*)dst, /*offset=*/0, /*cpol=*/0);
#else
    *(ushort8*)dst = *(const ushort8*)src;
#endif
  }
}

__device__ __forceinline__ void wait_async_lds() {
#if HAVE_ASYNC_LDS
#if __has_builtin(__builtin_amdgcn_s_wait_asynccnt)
  __builtin_amdgcn_s_wait_asynccnt(0);
#else
  asm volatile("s_wait_asynccnt 0" ::: "memory");
#endif
#endif
}

// ---------------------------------------------------------------------------
// Phase 1: one wave per 16-row M-tile; each wave holds all 13 N-tile
// accumulators so X streams from HBM exactly once (non-temporal, so Z
// stays L2-resident for phase 2). B chunks double-buffered in LDS.
// ---------------------------------------------------------------------------
__global__ __launch_bounds__(256) void snn_gemm_z(
    const float* __restrict__ x, const unsigned short* __restrict__ W0bf,
    const float* __restrict__ b0, float* __restrict__ Z) {
  __shared__ unsigned short ldsB[2][NROWS_PAD * LDSB_STRIDE];  // 2 x 16.6 KB

  const int tid  = threadIdx.x;
  const int wave = tid >> 5;
  const int lane = tid & 31;
  const int g    = lane >> 4;   // K-half-group selector (A/B layout)
  const int ln   = lane & 15;   // A row / B column / C column within tile

  const int  mtile = blockIdx.x * 8 + wave;
  const long mrow  = (long)mtile * 16 + ln;          // this lane's A row
  const float* __restrict__ xrow = x + mrow * SNN_F;

  v8f acc[NTILES];
#pragma unroll
  for (int i = 0; i < NTILES; ++i)
    acc[i] = (v8f){0.f, 0.f, 0.f, 0.f, 0.f, 0.f, 0.f, 0.f};

  // prologue: stage chunk 0 into buffer 0
  stage_b_chunk(W0bf, ldsB[0], 0, tid);
  wait_async_lds();
  __syncthreads();

  for (int kc = 0; kc < KCHUNKS; ++kc) {
    const int kbase = kc * 32;
    const int cur = kc & 1;

    // overlap: issue async copy of next chunk into the other buffer
    if (kc + 1 < KCHUNKS)
      stage_b_chunk(W0bf, ldsB[1 - cur], kbase + 32, tid);

    // ---- A fragment: 16-bit A 16x32 layout (ISA 7.12.2) ----
    // halves 0..7  <-> K = kbase + g*8 + {0..7}
    // halves 8..15 <-> K = kbase + 16 + g*8 + {0..7}
    union { unsigned short u[16]; v16bf v; } A;
#pragma unroll
    for (int half = 0; half < 2; ++half) {
      const int kb = kbase + half * 16 + g * 8;
#pragma unroll
      for (int q = 0; q < 8; q += 4) {
        v4f a = (v4f){0.f, 0.f, 0.f, 0.f};
        if (kb + q < SNN_F)  // k mult of 4, 784 % 4 == 0 -> all-or-nothing
          a = __builtin_nontemporal_load((const v4f*)(xrow + kb + q));
        A.u[half * 8 + q + 0] = f2bfbits(a.x);
        A.u[half * 8 + q + 1] = f2bfbits(a.y);
        A.u[half * 8 + q + 2] = f2bfbits(a.z);
        A.u[half * 8 + q + 3] = f2bfbits(a.w);
      }
    }

    // ---- 13 WMMAs sweep H; B fragments software-pipelined one ahead ----
    union Bu { ushort8 h[2]; v16bf v; };
    const unsigned short* bp = &ldsB[cur][ln * LDSB_STRIDE + g * 8];
    Bu bc, bn;
    bc.h[0] = *(const ushort8*)(bp);
    bc.h[1] = *(const ushort8*)(bp + 16);
#pragma unroll
    for (int nt = 0; nt < NTILES; ++nt) {
      if (nt + 1 < NTILES) {
        const unsigned short* q = bp + (nt + 1) * 16 * LDSB_STRIDE;
        bn.h[0] = *(const ushort8*)(q);
        bn.h[1] = *(const ushort8*)(q + 16);
      }
      acc[nt] = __builtin_amdgcn_wmma_f32_16x16x32_bf16(
          /*neg_a=*/false, A.v, /*neg_b=*/false, bc.v,
          /*c_mod=*/(short)0, acc[nt], /*reuse_a=*/false, /*reuse_b=*/false);
      bc = bn;
    }

    wait_async_lds();   // next chunk fully in LDS before anyone reads it
    __syncthreads();
  }

  // ---- epilogue: +bias, store Z (C layout: lane g*8+i rows, ln column) ----
  const long rowbase = (long)mtile * 16 + g * 8;
#pragma unroll
  for (int nt = 0; nt < NTILES; ++nt) {
    const int nn = nt * 16 + ln;
    if (nn < SNN_H) {
      const float bias = b0[nn];
#pragma unroll
      for (int i = 0; i < 8; ++i)
        Z[(rowbase + i) * SNN_H + nn] = acc[nt][i] + bias;
    }
  }
}

// ---------------------------------------------------------------------------
// Phase 2: one 64-thread block per batch element. Thread t owns hidden
// units 4t..4t+3 (t<50). Spike-weighted Wout partials reduced via LDS
// atomics; threads 0..9 advance the LI cell and write voltages.
// ---------------------------------------------------------------------------
__global__ __launch_bounds__(64) void snn_scan(
    const float* __restrict__ Z, const float* __restrict__ Wout,
    float* __restrict__ out) {
  __shared__ float red[SNN_O];

  const int b     = blockIdx.x;
  const int t     = threadIdx.x;
  const int hbase = t * 4;
  const bool active = (hbase < SNN_H);

  float wreg[SNN_O][4];
  if (active) {
#pragma unroll
    for (int o = 0; o < SNN_O; ++o)
#pragma unroll
      for (int j = 0; j < 4; ++j)
        wreg[o][j] = Wout[o * SNN_H + hbase + j];
  }

  float v[4]   = {0.f, 0.f, 0.f, 0.f};
  float cur[4] = {0.f, 0.f, 0.f, 0.f};
  float v_li = 0.f, i_li = 0.f;
  const float dtm = 0.1f;   // DT * tau_mem_inv
  const float kts = 0.8f;   // 1 - DT * tau_syn_inv

  for (int step = 0; step < SNN_T; ++step) {
    if (t < SNN_O) red[t] = 0.f;
    __syncthreads();

    if (active) {
      const v4f z4 = *(const v4f*)(Z + ((long)step * SNN_B + b) * SNN_H + hbase);
      const float zz[4] = {z4.x, z4.y, z4.z, z4.w};
      float part[SNN_O];
#pragma unroll
      for (int o = 0; o < SNN_O; ++o) part[o] = 0.f;
#pragma unroll
      for (int j = 0; j < 4; ++j) {
        const float vd = v[j] + dtm * (cur[j] - v[j]);  // v decay toward i
        const float id = cur[j] * kts;                  // synaptic decay
        const float sp = (vd > 1.0f) ? 1.f : 0.f;       // heaviside(v - Vth)
        v[j]   = (1.f - sp) * vd;                       // reset to 0 on spike
        cur[j] = id + zz[j];
#pragma unroll
        for (int o = 0; o < SNN_O; ++o) part[o] += sp * wreg[o][j];
      }
#pragma unroll
      for (int o = 0; o < SNN_O; ++o) atomicAdd(&red[o], part[o]);  // ds_add_f32
    }
    __syncthreads();

    if (t < SNN_O) {
      const float vnew = v_li + dtm * (i_li - v_li);   // uses OLD i_li
      i_li = i_li * kts + red[t];
      v_li = vnew;
      out[((long)step * SNN_B + b) * SNN_O + t] = vnew;
    }
  }
}

// ---------------------------------------------------------------------------
extern "C" void kernel_launch(void* const* d_in, const int* in_sizes, int n_in,
                              void* d_out, int out_size, void* d_ws, size_t ws_size,
                              hipStream_t stream) {
  (void)in_sizes; (void)n_in; (void)out_size; (void)ws_size;
  const float* x    = (const float*)d_in[0];   // [100,1024,784]
  const float* W0   = (const float*)d_in[1];   // [200,784]
  const float* b0   = (const float*)d_in[2];   // [200]
  const float* Wout = (const float*)d_in[3];   // [10,200]

  float* Z = (float*)d_ws;                              // 78.2 MB
  unsigned short* W0bf =
      (unsigned short*)((char*)d_ws + (size_t)MROWS * SNN_H * sizeof(float));

  snn_w0_prep<<<(NROWS_PAD * KPAD + 255) / 256, 256, 0, stream>>>(W0, W0bf);
  snn_gemm_z<<<MTILES / 8, 256, 0, stream>>>(x, W0bf, b0, Z);
  snn_scan<<<SNN_B, 64, 0, stream>>>(Z, Wout, (float*)d_out);
}
